// HeteroSAGELayer_85152021611239
// MI455X (gfx1250) — compile-verified, matrix operand order checked by reading, not run on hardware
//
#include <hip/hip_runtime.h>
#include <hip/hip_bf16.h>

#define N_EP   50000
#define N_FLOW 100000
#define EDGES  600000
#define D      128

typedef __attribute__((ext_vector_type(16))) _Float16 v16h;
typedef __attribute__((ext_vector_type(8)))  float    v8f;
typedef __attribute__((ext_vector_type(4)))  float    v4f;

// ---------------------------------------------------------------------------
// Zero a float region (grid-stride).
// ---------------------------------------------------------------------------
__global__ void zero_f32_kernel(float* __restrict__ p, size_t n) {
    size_t i = (size_t)blockIdx.x * blockDim.x + threadIdx.x;
    size_t stride = (size_t)gridDim.x * blockDim.x;
    for (; i < n; i += stride) p[i] = 0.0f;
}

// ---------------------------------------------------------------------------
// Pack one 128x128 f32 weight matrix (row-major, W[k][n]) into the per-lane
// WMMA B-fragment layout for v_wmma_f32_16x16x32_f16, converted to f16.
// Fragment (kb,nb): lane L -> column N = nb*16 + (L&15), khalf = (L>>4)*8;
//   halves 0..7  -> K = kb*32 + khalf + h
//   halves 8..15 -> K = kb*32 + khalf + 16 + (h-8)
// Packed element index i = ((kb*8+nb)*32 + lane)*16 + h (one v16h per lane).
// ---------------------------------------------------------------------------
__global__ void pack_weights_kernel(const float* __restrict__ W,
                                    _Float16* __restrict__ P) {
    int i = blockIdx.x * blockDim.x + threadIdx.x;
    if (i >= D * D) return;
    int h    = i & 15;
    int lane = (i >> 4) & 31;
    int frag = i >> 9;              // 0..31 = kb*8 + nb
    int kb   = frag >> 3;
    int nb   = frag & 7;
    int K = kb * 32 + ((lane >> 4) << 3) + (h < 8 ? h : h + 8);
    int N = nb * 16 + (lane & 15);
    P[i] = (_Float16)W[K * D + N];
}

// ---------------------------------------------------------------------------
// Edge scatter: one edge per wave32. Each lane atomically adds 4 consecutive
// features of x_src[src[e]] into sum[dst[e]]; lane 0 bumps the degree.
// Feature tables are L2-resident -> atomics resolve in the L2 atomic units.
// ---------------------------------------------------------------------------
__global__ __launch_bounds__(256) void scatter_sum_kernel(
    const float* __restrict__ x_src,
    const int*   __restrict__ src,
    const int*   __restrict__ dst,
    float*       __restrict__ sum,
    float*       __restrict__ deg,
    int num_edges)
{
    int e = blockIdx.x * 8 + (threadIdx.x >> 5);
    if (e >= num_edges) return;
    int lane = threadIdx.x & 31;
    int s = src[e];
    int t = dst[e];
    const v4f* xs = (const v4f*)(x_src + (size_t)s * D);
    v4f v = xs[lane];
    float* out = sum + (size_t)t * D + lane * 4;
    atomicAdd(out + 0, v.x);
    atomicAdd(out + 1, v.y);
    atomicAdd(out + 2, v.z);
    atomicAdd(out + 3, v.w);
    if (lane == 0) atomicAdd(deg + t, 1.0f);
}

// ---------------------------------------------------------------------------
// Fused SAGE GEMM:  Out = relu( X*Wself + (S/deg)*Wneigh + b )
// One block = 16 output rows; 8 waves = 8 N-tiles of 16 columns.
//
// Stage 1 (whole block): build the A_x and A_h f16 WMMA fragments for this
// row panel ONCE into LDS (invd folded into A_h during conversion).
// Stage 2 (per wave): 4x { 2 ds_load_b128 pairs + 2 packed-B global loads +
// 2 chained v_wmma_f32_16x16x32_f16 } into a single f32 accumulator.
//
// A-fragment layout in LDS: [kb][lane][16 halves] -> 32B-aligned v16h loads.
// ---------------------------------------------------------------------------
__global__ __launch_bounds__(256) void sage_gemm_kernel(
    const float*    __restrict__ X,      // [n, D] self features
    const float*    __restrict__ S,      // [n, D] neighbor sums
    const float*    __restrict__ deg,    // [n]
    const _Float16* __restrict__ WselfP, // packed B fragments
    const _Float16* __restrict__ WneighP,
    const float*    __restrict__ bias,   // [D]
    float*          __restrict__ Out,    // [n, D]
    int n)
{
    __shared__ _Float16 ldsX[4 * 32 * 16];   // A_x fragments (4 KB)
    __shared__ _Float16 ldsH[4 * 32 * 16];   // A_h fragments (4 KB)

    const int t     = threadIdx.x;
    const int mbase = blockIdx.x * 16;

    // ---- Stage A fragments into LDS (threads 0-127: X, 128-255: S/deg) ----
    {
        const int entry = t & 127;                 // kb*32 + lane
        const int kb    = entry >> 5;
        const int lane  = entry & 31;
        const int row   = mbase + (lane & 15);
        const int k0    = kb * 32 + ((lane >> 4) << 3);
        const bool isH  = (t >= 128);
        const float* base = (isH ? S : X) + (size_t)row * D + k0;
        const float scale =
            isH ? __builtin_amdgcn_rcpf(fmaxf(deg[row], 1.0f)) : 1.0f;
        v16h frag;
#pragma unroll
        for (int h = 0; h < 8; ++h) {
            frag[h]     = (_Float16)(base[h]      * scale);
            frag[h + 8] = (_Float16)(base[h + 16] * scale);
        }
        v16h* dst = (v16h*)((isH ? ldsH : ldsX) + entry * 16);
        *dst = frag;
    }
    __syncthreads();

    // ---- Per-wave WMMA main loop ----
    const int wave = t >> 5;        // 0..7 = N-tile
    const int lane = t & 31;
    const v16h* lx = (const v16h*)ldsX;
    const v16h* lh = (const v16h*)ldsH;
    const v16h* ws = (const v16h*)WselfP;
    const v16h* wn = (const v16h*)WneighP;

    v8f acc = {};
#pragma unroll
    for (int kb = 0; kb < 4; ++kb) {
        v16h ax = lx[kb * 32 + lane];
        v16h ah = lh[kb * 32 + lane];
        v16h bs = ws[(kb * 8 + wave) * 32 + lane];
        v16h bn = wn[(kb * 8 + wave) * 32 + lane];
        acc = __builtin_amdgcn_wmma_f32_16x16x32_f16(
                  false, ax, false, bs, (short)0, acc, false, false);
        acc = __builtin_amdgcn_wmma_f32_16x16x32_f16(
                  false, ah, false, bn, (short)0, acc, false, false);
    }

    // ---- Epilogue: bias + ReLU + store ----
    // C/D layout: VGPR r -> (M = r, N = lane) lanes 0-15,
    //             (M = r+8, N = lane-16) lanes 16-31.
    const int col   = wave * 16 + (lane & 15);
    const float bv  = bias[col];
    const int rbase = mbase + ((lane >> 4) << 3);
#pragma unroll
    for (int r = 0; r < 8; ++r) {
        float v = acc[r] + bv;
        Out[(size_t)(rbase + r) * D + col] = fmaxf(v, 0.0f);
    }
}

// ---------------------------------------------------------------------------
// Host-side launcher.
// Inputs (setup_inputs order):
//  0 x_endpoint [N_EP*D] f32      6 w_self_flow [D*D] f32
//  1 x_flow     [N_FLOW*D] f32    7 w_neigh_flow[D*D] f32
//  2 ef_src [E] i32               8 b_flow [D] f32
//  3 ef_dst [E] i32               9 w_self_ep  [D*D] f32
//  4 fe_src [E] i32              10 w_neigh_ep [D*D] f32
//  5 fe_dst [E] i32              11 b_ep   [D] f32
// Output: h_ep [N_EP*D] f32 then h_flow [N_FLOW*D] f32.
// ---------------------------------------------------------------------------
extern "C" void kernel_launch(void* const* d_in, const int* in_sizes, int n_in,
                              void* d_out, int out_size, void* d_ws, size_t ws_size,
                              hipStream_t stream) {
    const float* x_ep     = (const float*)d_in[0];
    const float* x_flow   = (const float*)d_in[1];
    const int*   ef_src   = (const int*)d_in[2];
    const int*   ef_dst   = (const int*)d_in[3];
    const int*   fe_src   = (const int*)d_in[4];
    const int*   fe_dst   = (const int*)d_in[5];
    const float* w_self_f = (const float*)d_in[6];
    const float* w_nei_f  = (const float*)d_in[7];
    const float* b_flow   = (const float*)d_in[8];
    const float* w_self_e = (const float*)d_in[9];
    const float* w_nei_e  = (const float*)d_in[10];
    const float* b_ep     = (const float*)d_in[11];

    float* h_ep_out   = (float*)d_out;                       // [N_EP*D]
    float* h_flow_out = (float*)d_out + (size_t)N_EP * D;    // [N_FLOW*D]

    // Workspace layout:
    char* ws = (char*)d_ws;
    const size_t PACKED_BYTES = (size_t)4 * D * D * sizeof(_Float16); // 128 KB
    _Float16* pW = (_Float16*)ws;                       // 4 packed matrices
    _Float16* p_self_flow = pW + 0 * D * D;
    _Float16* p_nei_flow  = pW + 1 * D * D;
    _Float16* p_self_ep   = pW + 2 * D * D;
    _Float16* p_nei_ep    = pW + 3 * D * D;

    float* sum_flow = (float*)(ws + PACKED_BYTES);                     // N_FLOW*D
    float* deg_flow = sum_flow + (size_t)N_FLOW * D;                   // N_FLOW
    float* sum_ep   = deg_flow + N_FLOW;                               // N_EP*D
    float* deg_ep   = sum_ep + (size_t)N_EP * D;                       // N_EP
    const size_t zero_elems = (size_t)N_FLOW * D + N_FLOW
                            + (size_t)N_EP * D + N_EP;

    // 1) Zero the accumulation scratch (sums + degrees, contiguous).
    zero_f32_kernel<<<2048, 256, 0, stream>>>(sum_flow, zero_elems);

    // 2) Pack the four weight matrices into WMMA B-fragment layout (f16).
    const int packGrid = (D * D + 255) / 256;
    pack_weights_kernel<<<packGrid, 256, 0, stream>>>(w_self_f, p_self_flow);
    pack_weights_kernel<<<packGrid, 256, 0, stream>>>(w_nei_f,  p_nei_flow);
    pack_weights_kernel<<<packGrid, 256, 0, stream>>>(w_self_e, p_self_ep);
    pack_weights_kernel<<<packGrid, 256, 0, stream>>>(w_nei_e,  p_nei_ep);

    // 3) Edge scatter-adds (one edge per wave, 8 edges per 256-thread block).
    const int scatGrid = (EDGES + 7) / 8;
    scatter_sum_kernel<<<scatGrid, 256, 0, stream>>>(
        x_ep, ef_src, ef_dst, sum_flow, deg_flow, EDGES);
    scatter_sum_kernel<<<scatGrid, 256, 0, stream>>>(
        x_flow, fe_src, fe_dst, sum_ep, deg_ep, EDGES);

    // 4) Fused WMMA GEMM + bias + ReLU per node type.
    sage_gemm_kernel<<<N_FLOW / 16, 256, 0, stream>>>(
        x_flow, sum_flow, deg_flow, p_self_flow, p_nei_flow, b_flow,
        h_flow_out, N_FLOW);
    sage_gemm_kernel<<<N_EP / 16, 256, 0, stream>>>(
        x_ep, sum_ep, deg_ep, p_self_ep, p_nei_ep, b_ep,
        h_ep_out, N_EP);
}
